// FDV_CL_37881611551170
// MI455X (gfx1250) — matrix operand order repaired and checked by reading, not compiled, not run on hardware
//
#include <hip/hip_runtime.h>
#include <math.h>

typedef __attribute__((ext_vector_type(16))) _Float16 v16h;
typedef __attribute__((ext_vector_type(8)))  _Float16 v8h;
typedef __attribute__((ext_vector_type(8)))  float    v8f;

#define B_ROWS   8192
#define M_DIM    256
#define COL_SEGS 8                   // column split of the 8192-wide LSE
#define SEG_COLS (B_ROWS / COL_SEGS) // 1024 columns per segment
#define SEG_TILES (SEG_COLS / 16)    // 64 16-col tiles per segment

__device__ __forceinline__ float waveReduceSum(float v) {
    v += __shfl_xor(v, 16, 32);
    v += __shfl_xor(v, 8, 32);
    v += __shfl_xor(v, 4, 32);
    v += __shfl_xor(v, 2, 32);
    v += __shfl_xor(v, 1, 32);
    return v;
}

// ---------------------------------------------------------------------------
// Kernel 1: SuffT[i][k] = sum_{j>=i} E[k][j]   (censoring suffix sums, 256KB)
// ---------------------------------------------------------------------------
__global__ void suffix_kernel(const float* __restrict__ E, float* __restrict__ SuffT) {
    const int i = blockIdx.x;    // suffix start (idx)
    const int k = threadIdx.x;   // output feature
    float s = 0.0f;
    for (int j = M_DIM - 1; j >= i; --j) s += E[k * M_DIM + j];
    SuffT[i * M_DIM + k] = s;
}

// ---------------------------------------------------------------------------
// Kernel 2: per-row prep. One wave32 per row:
//   time interp + censor mix -> y;  normalize z and y;  scale by 1/tau;
//   emit f16 hz/hy rows + fp32 diag_b = hz_b . hy_b
// ---------------------------------------------------------------------------
__global__ void __launch_bounds__(256) prep_kernel(
    const float* __restrict__ z, const float* __restrict__ t,
    const float* __restrict__ e, const float* __restrict__ log_tau,
    const float* __restrict__ E, const float* __restrict__ L,
    const float* __restrict__ SuffT,
    _Float16* __restrict__ hz_h, _Float16* __restrict__ hy_h,
    float* __restrict__ diag)
{
    const int wave = threadIdx.x >> 5;
    const int lane = threadIdx.x & 31;
    const int b    = blockIdx.x * 8 + wave;

    const float inv_tau = __expf(-0.5f * log_tau[0]);   // 1/sqrt(exp(log_tau))
    const float tb = t[b];
    const float eb = e[b];

    // searchsorted (lower_bound), then clamp to [1, m-1]
    int lo = 0, hi = M_DIM;
    while (lo < hi) { int mid = (lo + hi) >> 1; if (L[mid] < tb) lo = mid + 1; else hi = mid; }
    int idx = lo;
    if (idx == 0) idx = 1;
    if (idx == M_DIM) idx = M_DIM - 1;

    const float L_lo = L[idx - 1], L_hi = L[idx];
    const float tfrac   = (tb - L_lo) / (L_hi - L_lo);
    const float inv_den = 1.0f / (float)(M_DIM - idx);

    const float* __restrict__ Elo = E + (size_t)(idx - 1) * M_DIM;
    const float* __restrict__ Ehi = E + (size_t)idx * M_DIM;
    const float* __restrict__ St  = SuffT + (size_t)idx * M_DIM;

    float zv[8], yv[8];
    float sz = 0.0f, sy = 0.0f;
    #pragma unroll
    for (int u = 0; u < 8; ++u) {
        const int k = lane + 32 * u;
        const float zk  = z[(size_t)b * M_DIM + k];
        const float elo = Elo[k];
        const float yi  = elo + (Ehi[k] - elo) * tfrac;   // interp branch
        const float yc  = St[k] * inv_den;                // censor branch
        const float yk  = yi * eb + yc * (1.0f - eb);
        zv[u] = zk; yv[u] = yk;
        sz += zk * zk; sy += yk * yk;
    }
    sz = waveReduceSum(sz);
    sy = waveReduceSum(sy);
    const float scz = inv_tau / fmaxf(sqrtf(sz), 1e-12f);
    const float scy = inv_tau / fmaxf(sqrtf(sy), 1e-12f);

    float d = 0.0f;
    #pragma unroll
    for (int u = 0; u < 8; ++u) {
        const int k = lane + 32 * u;
        const float hz = zv[u] * scz;
        const float hy = yv[u] * scy;
        hz_h[(size_t)b * M_DIM + k] = (_Float16)hz;
        hy_h[(size_t)b * M_DIM + k] = (_Float16)hy;
        d += hz * hy;
    }
    d = waveReduceSum(d);
    if (lane == 0) diag[b] = d;
}

// ---------------------------------------------------------------------------
// Kernel 3: fused sim = hz @ hy^T (f16 WMMA, f32 acc) + partial exp-sums.
// Grid: 32 row-blocks x 8 column segments, 8 waves/block.
// Each wave owns TWO 16-row A tiles in registers (M=32): every B chunk
// loaded feeds 2 WMMAs, halving operand load traffic per FLOP.
// Fixed shift C = 1/tau^2 bounds |sim| -> exp(sim - C) needs no online max.
// partial[seg][row] = sum_{j in seg} exp(sim[row][j] - C)   (one writer each)
// ---------------------------------------------------------------------------
__global__ void __launch_bounds__(256) sim_lse_kernel(
    const _Float16* __restrict__ hz_h, const _Float16* __restrict__ hy_h,
    const float* __restrict__ log_tau, float* __restrict__ partial)
{
    const int wave     = threadIdx.x >> 5;
    const int lane     = threadIdx.x & 31;
    const int rowBlock = blockIdx.x >> 3;     // 0..31  (256 rows each)
    const int colSeg   = blockIdx.x & 7;      // 0..7   (1024 cols each)
    const int rowBase  = rowBlock * 256 + wave * 32;

    const int mrow  = lane & 15;
    const int khalf = lane >> 4;

    const float Cb = __expf(-log_tau[0]);     // 1/tau^2 >= |sim|

    // Preload two 16x256 A tiles in WMMA A layout:
    // lane halves [0..7] = K(kb + khalf*8 ..), halves [8..15] = K(kb+16+khalf*8 ..)
    v16h a[2][8];
    #pragma unroll
    for (int tI = 0; tI < 2; ++tI) {
        const _Float16* __restrict__ arow =
            hz_h + (size_t)(rowBase + tI * 16 + mrow) * M_DIM;
        #pragma unroll
        for (int k = 0; k < 8; ++k) {
            const v8h lo8 = *(const v8h*)(arow + 32 * k + khalf * 8);
            const v8h hi8 = *(const v8h*)(arow + 32 * k + 16 + khalf * 8);
            a[tI][k] = __builtin_shufflevector(lo8, hi8,
                           0,1,2,3,4,5,6,7,8,9,10,11,12,13,14,15);
        }
    }

    float s[2][8];
    #pragma unroll
    for (int tI = 0; tI < 2; ++tI)
        #pragma unroll
        for (int r = 0; r < 8; ++r) s[tI][r] = 0.0f;

    // B layout: lane holds column N = lane&15, K = khalf*8 + 0..7, 16 + khalf*8 + 0..7 (matches A)
    const _Float16* __restrict__ bbase =
        hy_h + (size_t)(colSeg * SEG_COLS + (lane & 15)) * M_DIM + khalf * 8;

    for (int j = 0; j < SEG_TILES; ++j) {
        const _Float16* __restrict__ bp = bbase + (size_t)j * (16 * M_DIM);
        v8f acc0 = {}, acc1 = {};
        #pragma unroll
        for (int k = 0; k < 8; ++k) {
            const v8h blo8 = *(const v8h*)(bp + 32 * k);
            const v8h bhi8 = *(const v8h*)(bp + 32 * k + 16);
            const v16h bmat = __builtin_shufflevector(blo8, bhi8,
                           0,1,2,3,4,5,6,7,8,9,10,11,12,13,14,15);
            acc0 = __builtin_amdgcn_wmma_f32_16x16x32_f16(
                       false, a[0][k], false, bmat, (short)0, acc0, false, false);
            acc1 = __builtin_amdgcn_wmma_f32_16x16x32_f16(
                       false, a[1][k], false, bmat, (short)0, acc1, false, false);
        }
        #pragma unroll
        for (int r = 0; r < 8; ++r) {
            s[0][r] += __expf(acc0[r] - Cb);
            s[1][r] += __expf(acc1[r] - Cb);
        }
    }

    // Reduce across the 16 lanes holding the same rows (masks < 16 keep the
    // two khalf groups separate under wave32).
    #pragma unroll
    for (int tI = 0; tI < 2; ++tI) {
        #pragma unroll
        for (int r = 0; r < 8; ++r) {
            float v = s[tI][r];
            v += __shfl_xor(v, 1, 32);
            v += __shfl_xor(v, 2, 32);
            v += __shfl_xor(v, 4, 32);
            v += __shfl_xor(v, 8, 32);
            s[tI][r] = v;
        }
    }

    if ((lane & 15) == 0) {
        float* __restrict__ pseg = partial + (size_t)colSeg * B_ROWS;
        #pragma unroll
        for (int tI = 0; tI < 2; ++tI) {
            #pragma unroll
            for (int r = 0; r < 8; ++r) {
                const int row = rowBase + tI * 16 + khalf * 8 + r; // C/D: M=r+8*khalf
                pseg[row] = s[tI][r];
            }
        }
    }
}

// ---------------------------------------------------------------------------
// Kernel 4: out[i] = clip(C + log(sum_seg partial) - diag[i] - log(B), -5, 15)
// ---------------------------------------------------------------------------
__global__ void finalize_kernel(const float* __restrict__ partial,
                                const float* __restrict__ diag,
                                const float* __restrict__ log_tau,
                                float* __restrict__ out)
{
    const int row = blockIdx.x * blockDim.x + threadIdx.x;
    float ssum = 0.0f;
    #pragma unroll
    for (int c = 0; c < COL_SEGS; ++c) ssum += partial[(size_t)c * B_ROWS + row];
    const float Cb   = __expf(-log_tau[0]);
    const float logB = logf((float)B_ROWS);
    float o = Cb + logf(ssum) - diag[row] - logB;
    out[row] = fminf(fmaxf(o, -5.0f), 15.0f);
}

extern "C" void kernel_launch(void* const* d_in, const int* in_sizes, int n_in,
                              void* d_out, int out_size, void* d_ws, size_t ws_size,
                              hipStream_t stream)
{
    (void)in_sizes; (void)n_in; (void)out_size; (void)ws_size;

    const float* z       = (const float*)d_in[0];
    const float* t       = (const float*)d_in[1];
    const float* e       = (const float*)d_in[2];
    const float* log_tau = (const float*)d_in[3];
    const float* E       = (const float*)d_in[4];  // time_emb_landmark [256,256]
    const float* L       = (const float*)d_in[5];  // time_landmark [256]
    float* out = (float*)d_out;

    // Workspace carve-out (~8.6 MB total)
    char* ws = (char*)d_ws;
    _Float16* hz_h  = (_Float16*)ws;  ws += (size_t)B_ROWS * M_DIM * sizeof(_Float16);
    _Float16* hy_h  = (_Float16*)ws;  ws += (size_t)B_ROWS * M_DIM * sizeof(_Float16);
    float*    SuffT = (float*)ws;     ws += (size_t)M_DIM * M_DIM * sizeof(float);
    float*    diag  = (float*)ws;     ws += (size_t)B_ROWS * sizeof(float);
    float*    part  = (float*)ws;     ws += (size_t)COL_SEGS * B_ROWS * sizeof(float);

    suffix_kernel<<<M_DIM, M_DIM, 0, stream>>>(E, SuffT);
    prep_kernel<<<B_ROWS / 8, 256, 0, stream>>>(z, t, e, log_tau, E, L, SuffT,
                                                hz_h, hy_h, diag);
    sim_lse_kernel<<<(B_ROWS / 256) * COL_SEGS, 256, 0, stream>>>(hz_h, hy_h,
                                                                  log_tau, part);
    finalize_kernel<<<B_ROWS / 256, 256, 0, stream>>>(part, diag, log_tau, out);
}
